// RIMCell_SharedParameters_76424648065328
// MI455X (gfx1250) — matrix operand order repaired and checked by reading, not compiled, not run on hardware
//
#include <hip/hip_runtime.h>
#include <hip/hip_bf16.h>
#include <cmath>

typedef __attribute__((ext_vector_type(16))) __bf16 v16bf;
typedef __attribute__((ext_vector_type(8)))  float  v8f;
typedef __attribute__((ext_vector_type(4)))  int    v4i;

#define TB  256
#define BM  128
#define BN  128
#define BK  32
#define LDK 40   // padded LDS row pitch in bf16 elems (80B) -> conflict-free b128 frag loads

__device__ __forceinline__ unsigned short f2bf(float f) {
    unsigned int u = __float_as_uint(f);
    u += 0x7FFFu + ((u >> 16) & 1u);     // round-to-nearest-even
    return (unsigned short)(u >> 16);
}

// ---- CDNA5 async global->LDS copy (ASYNCcnt) with safe fallback ------------
#if defined(__gfx1250__) && __has_builtin(__builtin_amdgcn_global_load_async_to_lds_b128)
#define HAVE_ASYNC_LDS 1
#endif

__device__ __forceinline__ void cp_b128(unsigned short* lds, const unsigned short* g) {
#ifdef HAVE_ASYNC_LDS
    // param0: int4 addrspace(1)* (global src) -- generic global ptrs are identity-mapped
    // param1: LDS dest -- low 32 bits of a generic shared ptr are the LDS byte offset (ISA aperture)
    __builtin_amdgcn_global_load_async_to_lds_b128(
        (__attribute__((address_space(1))) v4i*)(unsigned long long)g,
        (__attribute__((address_space(3))) v4i*)(unsigned int)(unsigned long long)lds,
        0, 0);
#else
    *(uint4*)lds = *(const uint4*)g;
#endif
}
__device__ __forceinline__ void async_wait() {
#ifdef HAVE_ASYNC_LDS
    asm volatile("s_wait_asynccnt 0" ::: "memory");
#endif
}

// ---------------------------------------------------------------------------
// bf16 WMMA GEMM: C[M,N](f32) = A[M,K](bf16) @ Wt[N,K](bf16, N-major).
// blockIdx.z = unit: Au = A + z*aoff (column offset, row stride arow),
// Wu = Wt + z*wstride, Cu = C + z*coff.
// epilogue: 0 = none, 1 = +bias[n], 2 = final (residual h_b, mask blend hs).
// ---------------------------------------------------------------------------
__global__ __launch_bounds__(TB) void gemm_bf16_wmma(
    const unsigned short* __restrict__ A, int arow, int aoff,
    const unsigned short* __restrict__ Wt, int wstride,
    float* __restrict__ C, int crow, int coff,
    int M, int N, int K,
    const float* __restrict__ bias, int epilogue,
    const float* __restrict__ hb, const float* __restrict__ hsin,
    const float* __restrict__ maskp, int mrow)
{
    __shared__ alignas(16) unsigned short As[2][BM * LDK];
    __shared__ alignas(16) unsigned short Bs[2][BN * LDK];

    const int u  = blockIdx.z;
    const unsigned short* Au = A  + (size_t)u * aoff;
    const unsigned short* Wu = Wt + (size_t)u * wstride;
    float*                Cu = C  + (size_t)u * coff;

    const int m0   = blockIdx.y * BM;
    const int n0   = blockIdx.x * BN;
    const int t    = threadIdx.x;
    const int lane = t & 31;
    const int wv   = t >> 5;      // wave id 0..7
    const int wm   = wv >> 2;     // 0..1  (64-row slab)
    const int wn   = wv & 3;      // 0..3  (32-col slab)

    // staging coords: each thread copies 16 bf16 (2 x b128) of one row
    const int rr  = t >> 1;            // 0..127 (row of A tile / row-n of B tile)
    const int seg = (t & 1) * 16;      // 0 or 16 (k offset)

    v8f acc[4][2];
    #pragma unroll
    for (int r = 0; r < 4; ++r)
        #pragma unroll
        for (int c = 0; c < 2; ++c)
            #pragma unroll
            for (int i = 0; i < 8; ++i) acc[r][c][i] = 0.f;

    const bool nFull = (n0 + BN <= N);

    auto stage = [&](int buf, int kk0) {
        const unsigned short* sa = Au + (size_t)(m0 + rr) * arow + kk0 + seg;
        unsigned short*       da = &As[buf][rr * LDK + seg];
        const unsigned short* sb = Wu + (size_t)(n0 + rr) * K + kk0 + seg;
        unsigned short*       db = &Bs[buf][rr * LDK + seg];
        if (kk0 + BK <= K) {               // uniform fast path (no guards)
            cp_b128(da, sa); cp_b128(da + 8, sa + 8);
            if (nFull) {
                cp_b128(db, sb); cp_b128(db + 8, sb + 8);
            } else {
                const bool rok = (n0 + rr) < N;
                uint4 z = make_uint4(0, 0, 0, 0);
                *(uint4*)db       = rok ? *(const uint4*)sb       : z;
                *(uint4*)(db + 8) = rok ? *(const uint4*)(sb + 8) : z;
            }
        } else {                           // ragged K tail (K % 8 == 0 always)
            uint4 z = make_uint4(0, 0, 0, 0);
            const int kc = kk0 + seg;
            *(uint4*)da       = (kc + 8  <= K) ? *(const uint4*)sa       : z;
            *(uint4*)(da + 8) = (kc + 16 <= K) ? *(const uint4*)(sa + 8) : z;
            const bool rok = (n0 + rr) < N;
            *(uint4*)db       = (rok && kc + 8  <= K) ? *(const uint4*)sb       : z;
            *(uint4*)(db + 8) = (rok && kc + 16 <= K) ? *(const uint4*)(sb + 8) : z;
        }
    };

    const int nk = (K + BK - 1) / BK;
    stage(0, 0);
    async_wait();
    __syncthreads();

    for (int s = 0; s < nk; ++s) {
        const int cur = s & 1;
        if (s + 1 < nk) stage(cur ^ 1, (s + 1) * BK);

        // Fragment loads matching ISA 16-bit A/B VGPR layouts (lane&15 = m/n,
        // lane>>4 = K-half; two contiguous ds_load_b128 per fragment).
        const int h16 = lane >> 4;
        v16bf afrag[4], bfrag[2];
        #pragma unroll
        for (int r = 0; r < 4; ++r) {
            const int row = wm * 64 + r * 16 + (lane & 15);
            union { uint4 q[2]; v16bf v; } uu;
            uu.q[0] = *(const uint4*)&As[cur][row * LDK + h16 * 8];
            uu.q[1] = *(const uint4*)&As[cur][row * LDK + 16 + h16 * 8];
            afrag[r] = uu.v;
        }
        #pragma unroll
        for (int c = 0; c < 2; ++c) {
            const int n = wn * 32 + c * 16 + (lane & 15);
            union { uint4 q[2]; v16bf v; } uu;
            uu.q[0] = *(const uint4*)&Bs[cur][n * LDK + h16 * 8];
            uu.q[1] = *(const uint4*)&Bs[cur][n * LDK + 16 + h16 * 8];
            bfrag[c] = uu.v;
        }
        #pragma unroll
        for (int r = 0; r < 4; ++r)
            #pragma unroll
            for (int c = 0; c < 2; ++c)
                acc[r][c] = __builtin_amdgcn_wmma_f32_16x16x32_bf16(
                    false, afrag[r], false, bfrag[c], (short)0, acc[r][c], false, false);

        async_wait();
        __syncthreads();
    }

    // Epilogue + store. C/D layout: elem i -> M = tile_m + i + 8*(lane>>4), N = tile_n + (lane&15)
    #pragma unroll
    for (int r = 0; r < 4; ++r) {
        const int tm = m0 + wm * 64 + r * 16 + (lane >> 4) * 8;
        #pragma unroll
        for (int c = 0; c < 2; ++c) {
            const int tn = n0 + wn * 32 + c * 16 + (lane & 15);
            if (tn >= N) continue;
            if (epilogue == 2) {
                const float* hbu = hb   + (size_t)u * coff;
                const float* hsu = hsin + (size_t)u * coff;
                #pragma unroll
                for (int i = 0; i < 8; ++i) {
                    const int b_ = tm + i;
                    const size_t idx = (size_t)b_ * crow + tn;
                    const float mk  = maskp[(size_t)b_ * mrow + u];
                    const float val = acc[r][c][i] + hbu[idx];
                    Cu[idx] = mk * val + (1.f - mk) * hsu[idx];
                }
            } else {
                const float badd = (epilogue == 1) ? bias[tn] : 0.f;
                #pragma unroll
                for (int i = 0; i < 8; ++i)
                    Cu[(size_t)(tm + i) * crow + tn] = acc[r][c][i] + badd;
            }
        }
    }
}

// ---------------------------------------------------------------------------
// Conversion / small fused kernels
// ---------------------------------------------------------------------------

// plain f32 -> bf16
__global__ void cvt_bf16(const float* __restrict__ in, unsigned short* __restrict__ out, long n)
{
    long idx = (long)blockIdx.x * blockDim.x + threadIdx.x;
    if (idx < n) out[idx] = f2bf(in[idx]);
}

// [z][k][n] f32 -> [z][n][k] bf16  (weights to N-major)
__global__ void cvt_bf16_t(const float* __restrict__ in, unsigned short* __restrict__ out,
                           int Kd, int Nd, long total)
{
    long idx = (long)blockIdx.x * blockDim.x + threadIdx.x;
    if (idx >= total) return;
    const long zo = idx / ((long)Kd * Nd);
    const long r  = idx % ((long)Kd * Nd);
    const int  nn = (int)(r / Kd), kk = (int)(r % Kd);
    out[idx] = f2bf(in[zo * Kd * Nd + (size_t)kk * Nd + nn]);
}

// Wt[u][n][k] = bf16( sum_s sw[u][s] * gw[s][k][n] )
__global__ void combine_w_t(const float* __restrict__ sw, const float* __restrict__ gw,
                            unsigned short* __restrict__ out, int Kd, int Nd)
{
    long idx = (long)blockIdx.x * blockDim.x + threadIdx.x;
    const long per = (long)Kd * Nd;
    if (idx >= 6 * per) return;
    const int  uu = (int)(idx / per);
    const long r  = idx % per;
    const int  nn = (int)(r / Kd), kk = (int)(r % Kd);
    float a = 0.f;
    #pragma unroll
    for (int s = 0; s < 4; ++s) a += sw[uu * 4 + s] * gw[(size_t)s * per + (size_t)kk * Nd + nn];
    out[idx] = f2bf(a);
}

// scores -> top4 mask + softmax-over-2-steps prob (step1 key = bk)
__global__ void score_mask(const float* __restrict__ ql, const float* __restrict__ k0,
                           const float* __restrict__ bk,
                           float* __restrict__ p, float* __restrict__ mask, int B)
{
    const int b = blockIdx.x * blockDim.x + threadIdx.x;
    if (b >= B) return;
    float s0[6], s1[6];
    for (int uu = 0; uu < 6; ++uu) {
        float a = 0.f, c = 0.f;
        const float* q = ql + ((size_t)b * 6 + uu) * 64;
        #pragma unroll 8
        for (int j = 0; j < 64; ++j) { a += q[j] * k0[(size_t)b * 64 + j]; c += q[j] * bk[j]; }
        s0[uu] = a * 0.125f; s1[uu] = c * 0.125f;
    }
    for (int uu = 0; uu < 6; ++uu) {
        int rank = 0;
        for (int u2 = 0; u2 < 6; ++u2)
            rank += (s0[u2] > s0[uu]) || (s0[u2] == s0[uu] && u2 < uu);
        mask[(size_t)b * 6 + uu] = (rank < 4) ? 1.f : 0.f;
        p[(size_t)b * 6 + uu] = 1.f / (1.f + __expf(s1[uu] - s0[uu]));   // sigmoid(s0-s1)
    }
}

// inputs[b,u,j] = bf16( mask * (p*v0[b,j] + (1-p)*bv[j]) )
__global__ void make_inputs(const float* __restrict__ p, const float* __restrict__ mask,
                            const float* __restrict__ v0, const float* __restrict__ bv,
                            unsigned short* __restrict__ inp, long total)
{
    long idx = (long)blockIdx.x * blockDim.x + threadIdx.x;
    if (idx >= total) return;
    const long b = idx / 2400;
    const int  r = (int)(idx % 2400);
    const int  uu = r / 400, j = r % 400;
    const float p0 = p[b * 6 + uu];
    inp[idx] = f2bf(mask[b * 6 + uu] * (p0 * v0[b * 400 + j] + (1.f - p0) * bv[j]));
}

// GRU gates -> h_b (f32 for residual/epilogue + bf16 for GEMMs)
__global__ void gru_fuse(const float* __restrict__ gx, const float* __restrict__ gh,
                         const float* __restrict__ hs, float* __restrict__ hb,
                         unsigned short* __restrict__ hbbf, long total)
{
    long idx = (long)blockIdx.x * blockDim.x + threadIdx.x;
    if (idx >= total) return;
    const long bu = idx >> 9;
    const int  n  = (int)(idx & 511);
    const float* gxb = gx + bu * 1536;
    const float* ghb = gh + bu * 1536;
    const float rg = 1.f / (1.f + __expf(-(gxb[n]       + ghb[n])));
    const float zg = 1.f / (1.f + __expf(-(gxb[512 + n] + ghb[512 + n])));
    const float ng = tanhf(gxb[1024 + n] + rg * ghb[1024 + n]);
    const float h  = hs[idx];
    const float o  = ng + zg * (h - ng);
    hb[idx]   = o;
    hbbf[idx] = f2bf(o);
}

// per (b,head): 6x6 attention softmax * query-mask -> ap
__global__ void attn_probs(const float* __restrict__ q, const float* __restrict__ kk,
                           const float* __restrict__ mask, float* __restrict__ ap, int B)
{
    const int idx = blockIdx.x * blockDim.x + threadIdx.x;
    if (idx >= B * 4) return;
    const int b = idx >> 2, h = idx & 3;
    const float scale = 0.17677669529663687f;  // 1/sqrt(32)
    float att[6][6];
    for (int qi = 0; qi < 6; ++qi) {
        const float* qp = q + ((size_t)b * 6 + qi) * 128 + h * 32;
        for (int ki = 0; ki < 6; ++ki) {
            const float* kp = kk + ((size_t)b * 6 + ki) * 128 + h * 32;
            float a = 0.f;
            #pragma unroll 8
            for (int d = 0; d < 32; ++d) a += qp[d] * kp[d];
            att[qi][ki] = a * scale;
        }
    }
    float* apb = ap + (size_t)idx * 36;
    for (int qi = 0; qi < 6; ++qi) {
        float mx = att[qi][0];
        for (int ki = 1; ki < 6; ++ki) mx = fmaxf(mx, att[qi][ki]);
        float den = 0.f, e[6];
        for (int ki = 0; ki < 6; ++ki) { e[ki] = __expf(att[qi][ki] - mx); den += e[ki]; }
        const float mk = mask[(size_t)b * 6 + qi] / den;
        for (int ki = 0; ki < 6; ++ki) apb[qi * 6 + ki] = e[ki] * mk;
    }
}

// ctx2[b,u, h*512+d] = bf16( sum_k ap[b,h,u,k] * v[b,k, h*512+d] )
__global__ void ctx_mix(const float* __restrict__ ap, const float* __restrict__ v,
                        unsigned short* __restrict__ ctx, long total)
{
    long idx = (long)blockIdx.x * blockDim.x + threadIdx.x;
    if (idx >= total) return;
    const long b = idx >> 11;
    const int  j = (int)(idx & 2047);
    const int  h = j >> 9;
    const float* apb = ap + ((size_t)b * 4 + h) * 36;
    float a[6] = {0.f, 0.f, 0.f, 0.f, 0.f, 0.f};
    #pragma unroll
    for (int k = 0; k < 6; ++k) {
        const float vv = v[((size_t)b * 6 + k) * 2048 + j];
        #pragma unroll
        for (int uu = 0; uu < 6; ++uu) a[uu] += apb[uu * 6 + k] * vv;
    }
    #pragma unroll
    for (int uu = 0; uu < 6; ++uu) ctx[((size_t)b * 6 + uu) * 2048 + j] = f2bf(a[uu]);
}

// ---------------------------------------------------------------------------
extern "C" void kernel_launch(void* const* d_in, const int* in_sizes, int n_in,
                              void* d_out, int out_size, void* d_ws, size_t ws_size,
                              hipStream_t stream)
{
    const float* x      = (const float*)d_in[0];
    const float* hs     = (const float*)d_in[1];
    const float* Wk     = (const float*)d_in[2];
    const float* bk     = (const float*)d_in[3];
    const float* Wv     = (const float*)d_in[4];
    const float* bv     = (const float*)d_in[5];
    const float* Wq     = (const float*)d_in[6];
    const float* Wq_    = (const float*)d_in[7];
    const float* Wk_    = (const float*)d_in[8];
    const float* Wv_    = (const float*)d_in[9];
    const float* Wo_    = (const float*)d_in[10];
    const float* sw     = (const float*)d_in[11];
    const float* gru_wx = (const float*)d_in[12];
    const float* gru_wh = (const float*)d_in[13];
    float* out = (float*)d_out;

    const int B = in_sizes[0] / 1024;           // 4096

    // ---- workspace carve-up (byte allocator, 256B aligned) ----
    char* base = (char*)d_ws;
    size_t off = 0;
    auto alloc = [&](size_t bytes) { off = (off + 255) & ~(size_t)255; void* p = base + off; off += bytes; return p; };
    auto fa = [&](size_t n) { return (float*)alloc(n * 4); };
    auto ha = [&](size_t n) { return (unsigned short*)alloc(n * 2); };

    // f32 intermediates
    float* k0    = fa((size_t)B * 64);
    float* v0    = fa((size_t)B * 400);
    float* ql    = fa((size_t)B * 6 * 64);
    float* pbuf  = fa((size_t)B * 6);
    float* maskb = fa((size_t)B * 6);
    float* gx    = fa((size_t)B * 6 * 1536);
    float* gh    = fa((size_t)B * 6 * 1536);
    float* hb    = fa((size_t)B * 6 * 512);
    float* qb    = fa((size_t)B * 6 * 128);
    float* kb    = fa((size_t)B * 6 * 128);
    float* apb   = fa((size_t)B * 4 * 36);
    float* vbuf  = gx;   // alias: gx+gh dead after GRU; v needs B*12288 <= B*18432 span

    // bf16 operands
    unsigned short* xbf   = ha((size_t)B * 1024);
    unsigned short* hsbf  = ha((size_t)B * 3072);
    unsigned short* inbf  = ha((size_t)B * 2400);
    unsigned short* hbbf  = ha((size_t)B * 3072);
    unsigned short* c2bf  = ha((size_t)B * 12288);
    unsigned short* Wk_t  = ha((size_t)64 * 1024);
    unsigned short* Wv_t  = ha((size_t)400 * 1024);
    unsigned short* Wq_t  = ha((size_t)6 * 64 * 512);
    unsigned short* Wq2_t = ha((size_t)6 * 128 * 512);
    unsigned short* Wk2_t = ha((size_t)6 * 128 * 512);
    unsigned short* Wv2_t = ha((size_t)6 * 2048 * 512);
    unsigned short* Wo2_t = ha((size_t)6 * 512 * 2048);
    unsigned short* Wx_t  = ha((size_t)6 * 1536 * 400);
    unsigned short* Wh_t  = ha((size_t)6 * 1536 * 512);

    const dim3 blk(TB);
    const int MY = B / BM;
    auto cdiv = [](long a, long b) { return (int)((a + b - 1) / b); };

    // 0) one-time bf16 conversions (activations row-major, weights N-major)
    cvt_bf16<<<cdiv((long)B * 1024, TB), blk, 0, stream>>>(x,  xbf,  (long)B * 1024);
    cvt_bf16<<<cdiv((long)B * 3072, TB), blk, 0, stream>>>(hs, hsbf, (long)B * 3072);
    cvt_bf16_t<<<cdiv((long)1024 * 64,  TB), blk, 0, stream>>>(Wk,  Wk_t,  1024, 64,   (long)1024 * 64);
    cvt_bf16_t<<<cdiv((long)1024 * 400, TB), blk, 0, stream>>>(Wv,  Wv_t,  1024, 400,  (long)1024 * 400);
    cvt_bf16_t<<<cdiv((long)6 * 512 * 64,   TB), blk, 0, stream>>>(Wq,  Wq_t,  512, 64,   (long)6 * 512 * 64);
    cvt_bf16_t<<<cdiv((long)6 * 512 * 128,  TB), blk, 0, stream>>>(Wq_, Wq2_t, 512, 128,  (long)6 * 512 * 128);
    cvt_bf16_t<<<cdiv((long)6 * 512 * 128,  TB), blk, 0, stream>>>(Wk_, Wk2_t, 512, 128,  (long)6 * 512 * 128);
    cvt_bf16_t<<<cdiv((long)6 * 512 * 2048, TB), blk, 0, stream>>>(Wv_, Wv2_t, 512, 2048, (long)6 * 512 * 2048);
    cvt_bf16_t<<<cdiv((long)6 * 2048 * 512, TB), blk, 0, stream>>>(Wo_, Wo2_t, 2048, 512, (long)6 * 2048 * 512);
    combine_w_t<<<cdiv((long)6 * 400 * 1536, TB), blk, 0, stream>>>(sw, gru_wx, Wx_t, 400, 1536);
    combine_w_t<<<cdiv((long)6 * 512 * 1536, TB), blk, 0, stream>>>(sw, gru_wh, Wh_t, 512, 1536);

    // 1) k0 = x@Wk + bk ; v0 = x@Wv + bv
    gemm_bf16_wmma<<<dim3(1, MY, 1), blk, 0, stream>>>(xbf, 1024, 0, Wk_t, 0, k0, 64, 0,
        B, 64, 1024, bk, 1, nullptr, nullptr, nullptr, 0);
    gemm_bf16_wmma<<<dim3(4, MY, 1), blk, 0, stream>>>(xbf, 1024, 0, Wv_t, 0, v0, 400, 0,
        B, 400, 1024, bv, 1, nullptr, nullptr, nullptr, 0);

    // 2) q_l[u] = hs[:,u,:] @ Wq[u]
    gemm_bf16_wmma<<<dim3(1, MY, 6), blk, 0, stream>>>(hsbf, 3072, 512, Wq_t, 64 * 512, ql, 384, 64,
        B, 64, 512, nullptr, 0, nullptr, nullptr, nullptr, 0);

    // 3) scores -> mask + probs ; 4) inputs (bf16)
    score_mask<<<cdiv(B, TB), blk, 0, stream>>>(ql, k0, bk, pbuf, maskb, B);
    make_inputs<<<cdiv((long)B * 6 * 400, TB), blk, 0, stream>>>(pbuf, maskb, v0, bv, inbf, (long)B * 6 * 400);

    // 5) gx[u] = inputs[:,u,:] @ Wx[u] ; gh[u] = hs[:,u,:] @ Wh[u]
    gemm_bf16_wmma<<<dim3(12, MY, 6), blk, 0, stream>>>(inbf, 2400, 400, Wx_t, 1536 * 400, gx, 9216, 1536,
        B, 1536, 400, nullptr, 0, nullptr, nullptr, nullptr, 0);
    gemm_bf16_wmma<<<dim3(12, MY, 6), blk, 0, stream>>>(hsbf, 3072, 512, Wh_t, 1536 * 512, gh, 9216, 1536,
        B, 1536, 512, nullptr, 0, nullptr, nullptr, nullptr, 0);

    // 6) GRU -> h_b (f32 + bf16)
    gru_fuse<<<cdiv((long)B * 6 * 512, TB), blk, 0, stream>>>(gx, gh, hs, hb, hbbf, (long)B * 6 * 512);

    // 7) q/k/v projections from h_b
    gemm_bf16_wmma<<<dim3(1, MY, 6), blk, 0, stream>>>(hbbf, 3072, 512, Wq2_t, 128 * 512, qb, 768, 128,
        B, 128, 512, nullptr, 0, nullptr, nullptr, nullptr, 0);
    gemm_bf16_wmma<<<dim3(1, MY, 6), blk, 0, stream>>>(hbbf, 3072, 512, Wk2_t, 128 * 512, kb, 768, 128,
        B, 128, 512, nullptr, 0, nullptr, nullptr, nullptr, 0);
    gemm_bf16_wmma<<<dim3(16, MY, 6), blk, 0, stream>>>(hbbf, 3072, 512, Wv2_t, 2048 * 512, vbuf, 12288, 2048,
        B, 2048, 512, nullptr, 0, nullptr, nullptr, nullptr, 0);

    // 8) 6x6 attention probs ; 9) context mix (bf16 ctx2)
    attn_probs<<<cdiv(B * 4, TB), blk, 0, stream>>>(qb, kb, maskb, apb, B);
    ctx_mix<<<cdiv((long)B * 2048, TB), blk, 0, stream>>>(apb, vbuf, c2bf, (long)B * 2048);

    // 10) out[u] = mask*(ctx2[u]@Wo_[u] + h_b) + (1-mask)*hs  -> d_out
    gemm_bf16_wmma<<<dim3(4, MY, 6), blk, 0, stream>>>(c2bf, 12288, 2048, Wo2_t, 512 * 2048, out, 3072, 512,
        B, 512, 2048, nullptr, 2, hb, hs, maskb, 6);
}
